// CCN1D_7584912245325
// MI455X (gfx1250) — matrix operand order repaired and checked by reading, not compiled, not run on hardware
//
#include <hip/hip_runtime.h>
#include <hip/hip_bf16.h>

typedef __attribute__((ext_vector_type(16))) __bf16        v16bf;
typedef __attribute__((ext_vector_type(8)))  float         v8f;
typedef __attribute__((ext_vector_type(8)))  unsigned int  v8u;

#define N_VERT 50000
#define KP1    17
#define N_MOLS 500
#define VPM    100   // vertices per molecule

// ---------- bf16 helpers ----------
__device__ __forceinline__ unsigned int f2bf_bits(float f) {
  unsigned int u = __builtin_bit_cast(unsigned int, f);
  u += 0x7fffu + ((u >> 16) & 1u);           // round-to-nearest-even
  return u >> 16;
}
__device__ __forceinline__ unsigned int pack2bf(float lo, float hi) {
  return f2bf_bits(lo) | (f2bf_bits(hi) << 16);
}

// packed bf16 (add, then relu) entirely in VOP3P — 2 VALU ops, no unpack/repack
__device__ __forceinline__ unsigned int pk_add_relu_bf16(unsigned int a, unsigned int b) {
  unsigned int s, r;
  asm("v_pk_add_bf16 %0, %1, %2" : "=v"(s) : "v"(a), "v"(b));
  asm("v_pk_max_num_bf16 %0, %1, 0" : "=v"(r) : "v"(s));
  return r;
}

// ---------- weight packing into WMMA B-fragment layout ----------
// B (32x16 bf16) per-lane layout: col = lane&15; lanes 0-15 hold K=0..7,16..23,
// lanes 16-31 hold K=8..15,24..31; dword d packs K=8h+2(d&3)+16(d>>2) and +1.
// Output: frag[(tile*ksteps + ks)*32 + lane][8 dwords]
__global__ void pack_b_frags(const float* __restrict__ W, int wcols, int row_base,
                             int ksteps, int ntiles, float scale,
                             unsigned int* __restrict__ out) {
  int tid = blockIdx.x * blockDim.x + threadIdx.x;
  int total = ntiles * ksteps * 32 * 8;
  if (tid >= total) return;
  int d    = tid & 7;
  int lane = (tid >> 3) & 31;
  int rest = tid >> 8;
  int ks   = rest % ksteps;
  int tile = rest / ksteps;
  int col  = tile * 16 + (lane & 15);
  int k0   = ks * 32 + ((lane >= 16) ? 8 : 0) + (d & 3) * 2 + (d >> 2) * 16;
  int r0   = row_base + k0;
  float lo = W[r0 * wcols + col] * scale;
  float hi = W[(r0 + 1) * wcols + col] * scale;
  out[((tile * ksteps + ks) * 32 + lane) * 8 + d] = pack2bf(lo, hi);
}

// ---------- F0 = relu(embed[x]); also bf16 copy ----------
__global__ void embed_kernel(const int* __restrict__ x, const float* __restrict__ embed,
                             float* __restrict__ F0, unsigned short* __restrict__ Sbf) {
  int i = blockIdx.x * 256 + threadIdx.x;
  if (i >= N_VERT * 64) return;
  int v = i >> 6, d = i & 63;
  float f = fmaxf(embed[x[v] * 64 + d], 0.f);
  F0[i]  = f;
  Sbf[i] = (unsigned short)f2bf_bits(f);
}

// ---------- dense expand:  Gt = S @ (Wa_top/17),  Gb = S @ (Wa_bot/17) ----------
// Sbf: [N,64] bf16 row-major.  packA: 16 tiles (0-7 -> Gt cols, 8-15 -> Gb cols) x 2 ksteps.
__global__ void __launch_bounds__(256)
gemm_expand(const unsigned short* __restrict__ Sbf,
            const unsigned int*  __restrict__ packA,
            unsigned short* __restrict__ Gt,
            unsigned short* __restrict__ Gb, int n) {
  int lane  = threadIdx.x & 31;
  int wave  = threadIdx.x >> 5;
  int vbase = (blockIdx.x * 8 + wave) * 16;
  if (vbase >= n) return;
  int mrow = lane & 15;
  int kb   = (lane >= 16) ? 8 : 0;   // A layout: half-runs of 8 consecutive K

  // A fragments, K = 64 -> 2 ksteps.  Data already bf16 in memory order.
  v16bf a[2];
  const unsigned short* srow = Sbf + (vbase + mrow) * 64;
#pragma unroll
  for (int ks = 0; ks < 2; ++ks) {
    uint4 r0 = *reinterpret_cast<const uint4*>(srow + ks * 32 + kb);
    uint4 r1 = *reinterpret_cast<const uint4*>(srow + ks * 32 + kb + 16);
    v8u p; p[0]=r0.x; p[1]=r0.y; p[2]=r0.z; p[3]=r0.w; p[4]=r1.x; p[5]=r1.y; p[6]=r1.z; p[7]=r1.w;
    a[ks] = __builtin_bit_cast(v16bf, p);
  }

  int rb = vbase + ((lane >= 16) ? 8 : 0);   // D rows: M = i + 8*half
#pragma unroll
  for (int t = 0; t < 16; ++t) {
    v8f c = {};
#pragma unroll
    for (int ks = 0; ks < 2; ++ks) {
      const unsigned int* bp = packA + ((t * 2 + ks) * 32 + lane) * 8;
      uint4 b0 = *reinterpret_cast<const uint4*>(bp);
      uint4 b1 = *reinterpret_cast<const uint4*>(bp + 4);
      v8u p; p[0]=b0.x; p[1]=b0.y; p[2]=b0.z; p[3]=b0.w; p[4]=b1.x; p[5]=b1.y; p[6]=b1.z; p[7]=b1.w;
      c = __builtin_amdgcn_wmma_f32_16x16x32_bf16(false, a[ks], false,
            __builtin_bit_cast(v16bf, p), (short)0, c, false, false);
    }
    unsigned short* dst = (t < 8) ? Gt : Gb;
    int colb = (t & 7) * 16 + (lane & 15);
#pragma unroll
    for (int i = 0; i < 8; ++i)
      dst[(rb + i) * 128 + colb] = (unsigned short)f2bf_bits(c[i]);
  }
}

// ---------- main fused layer kernel ----------
// Per block: 16 vertices -> 272 RF rows -> 17 M-tiles.
// Row r: A-row = relu(Gt[nbr(r)] + Gb[v(r)]) (bf16) ; x Wb[128,64] ; relu ; sum over 17.
__global__ void __launch_bounds__(256)
ccn_layer_kernel(const int* __restrict__ rf,
                 const unsigned short* __restrict__ Gt,
                 const unsigned short* __restrict__ Gb,
                 const unsigned int*  __restrict__ packB,
                 float* __restrict__ s_out,
                 unsigned short* __restrict__ s_out_bf) {
  __shared__ float acc[16 * 64];
  int tid = threadIdx.x;
  for (int e = tid; e < 16 * 64; e += 256) acc[e] = 0.f;
  __syncthreads();

  int lane  = tid & 31;
  int wave  = tid >> 5;
  int vbase = blockIdx.x * 16;
  int mrow  = lane & 15;
  int half  = (lane >= 16) ? 1 : 0;
  int kb    = half * 8;

  for (int tile = wave; tile < KP1; tile += 8) {
    // gather + packed-bf16 add/relu fused into A-fragment build
    int lr  = tile * 16 + mrow;
    int lv  = lr / KP1;
    int kr  = lr - lv * KP1;
    int v   = vbase + lv;
    int nbr = rf[v * KP1 + kr];
    const unsigned short* gtr = Gt + nbr * 128;
    const unsigned short* gbr = Gb + v * 128;

    v16bf a[4];
#pragma unroll
    for (int ks = 0; ks < 4; ++ks) {
      int o0 = ks * 32 + kb;
      uint4 g0 = *reinterpret_cast<const uint4*>(gtr + o0);
      uint4 h0 = *reinterpret_cast<const uint4*>(gbr + o0);
      uint4 g1 = *reinterpret_cast<const uint4*>(gtr + o0 + 16);
      uint4 h1 = *reinterpret_cast<const uint4*>(gbr + o0 + 16);
      v8u p;
      p[0] = pk_add_relu_bf16(g0.x, h0.x);
      p[1] = pk_add_relu_bf16(g0.y, h0.y);
      p[2] = pk_add_relu_bf16(g0.z, h0.z);
      p[3] = pk_add_relu_bf16(g0.w, h0.w);
      p[4] = pk_add_relu_bf16(g1.x, h1.x);
      p[5] = pk_add_relu_bf16(g1.y, h1.y);
      p[6] = pk_add_relu_bf16(g1.z, h1.z);
      p[7] = pk_add_relu_bf16(g1.w, h1.w);
      a[ks] = __builtin_bit_cast(v16bf, p);
    }

    // per-row LDS accumulator bases (hoisted: 8 divisions per tile, not 32)
    float* accp[8];
#pragma unroll
    for (int i = 0; i < 8; ++i) {
      int lr2 = tile * 16 + i + half * 8;
      accp[i] = &acc[(lr2 / KP1) * 64 + (lane & 15)];
    }

#pragma unroll
    for (int nt = 0; nt < 4; ++nt) {
      v8f c = {};
#pragma unroll
      for (int ks = 0; ks < 4; ++ks) {
        const unsigned int* bp = packB + ((nt * 4 + ks) * 32 + lane) * 8;
        uint4 b0 = *reinterpret_cast<const uint4*>(bp);
        uint4 b1 = *reinterpret_cast<const uint4*>(bp + 4);
        v8u pb; pb[0]=b0.x; pb[1]=b0.y; pb[2]=b0.z; pb[3]=b0.w; pb[4]=b1.x; pb[5]=b1.y; pb[6]=b1.z; pb[7]=b1.w;
        c = __builtin_amdgcn_wmma_f32_16x16x32_bf16(false, a[ks], false,
              __builtin_bit_cast(v16bf, pb), (short)0, c, false, false);
      }
#pragma unroll
      for (int i = 0; i < 8; ++i)
        atomicAdd(accp[i] + nt * 16, fmaxf(c[i], 0.f));   // relu then segment-sum
    }
  }
  __syncthreads();

  for (int e = tid; e < 16 * 64; e += 256) {
    int v = vbase + (e >> 6);
    float val = acc[e];
    s_out[v * 64 + (e & 63)]    = val;
    s_out_bf[v * 64 + (e & 63)] = (unsigned short)f2bf_bits(val);  // next layer input
  }
}

// ---------- molecule pooling + FC ----------
__global__ void mol_kernel(const float* __restrict__ F0, const float* __restrict__ s0,
                           const float* __restrict__ s1, const float* __restrict__ fc_w,
                           const float* __restrict__ fc_b, float* __restrict__ out) {
  __shared__ float rep[192];
  int m = blockIdx.x;
  int t = threadIdx.x;                           // 192 threads
  const float* src = (t < 64) ? F0 : ((t < 128) ? s0 : s1);
  int col = t & 63;
  int base = m * VPM * 64;
  float s = 0.f;
  for (int r = 0; r < VPM; ++r) s += src[base + r * 64 + col];
  rep[t] = s;
  __syncthreads();
  if (t < 32) {
    float o = fc_b[t];
    for (int c = 0; c < 192; ++c) o += rep[c] * fc_w[c * 32 + t];
    out[m * 32 + t] = o;
  }
}

// ---------- orchestration ----------
extern "C" void kernel_launch(void* const* d_in, const int* in_sizes, int n_in,
                              void* d_out, int out_size, void* d_ws, size_t ws_size,
                              hipStream_t stream) {
  const int*   x     = (const int*)d_in[0];
  const int*   rf    = (const int*)d_in[1];
  // d_in[2] = mol_ids: molecules are contiguous blocks of 100 vertices by construction
  const float* embed = (const float*)d_in[3];
  const float* W0_0  = (const float*)d_in[4];
  const float* W0_1  = (const float*)d_in[5];
  const float* W1_0  = (const float*)d_in[6];
  const float* W1_1  = (const float*)d_in[7];
  const float* fc_w  = (const float*)d_in[8];
  const float* fc_b  = (const float*)d_in[9];
  float* out = (float*)d_out;

  char* ws = (char*)d_ws;
  float*          F0     = (float*)(ws);                 ws += (size_t)N_VERT * 64 * 4;
  unsigned short* Sbf    = (unsigned short*)(ws);        ws += (size_t)N_VERT * 64 * 2;
  unsigned short* Gt     = (unsigned short*)(ws);        ws += (size_t)N_VERT * 128 * 2;
  unsigned short* Gb     = (unsigned short*)(ws);        ws += (size_t)N_VERT * 128 * 2;
  float*          s0     = (float*)(ws);                 ws += (size_t)N_VERT * 64 * 4;
  float*          s1     = (float*)(ws);                 ws += (size_t)N_VERT * 64 * 4;
  unsigned int*   packA0 = (unsigned int*)(ws);          ws += 16 * 2 * 32 * 8 * 4;
  unsigned int*   packA1 = (unsigned int*)(ws);          ws += 16 * 2 * 32 * 8 * 4;
  unsigned int*   packB0 = (unsigned int*)(ws);          ws += 4 * 4 * 32 * 8 * 4;
  unsigned int*   packB1 = (unsigned int*)(ws);          ws += 4 * 4 * 32 * 8 * 4;

  const float inv17 = 1.f / 17.f;
  // pack weights (scale 1/17 folded into Wa; tiles 0-7 = rows 0-63 -> Gt, 8-15 = rows 64-127 -> Gb)
  pack_b_frags<<<16, 256, 0, stream>>>(W0_0, 128, 0,  2, 8, inv17, packA0);
  pack_b_frags<<<16, 256, 0, stream>>>(W0_0, 128, 64, 2, 8, inv17, packA0 + 8 * 2 * 32 * 8);
  pack_b_frags<<<16, 256, 0, stream>>>(W0_1, 64,  0,  4, 4, 1.f,   packB0);
  pack_b_frags<<<16, 256, 0, stream>>>(W1_0, 128, 0,  2, 8, inv17, packA1);
  pack_b_frags<<<16, 256, 0, stream>>>(W1_0, 128, 64, 2, 8, inv17, packA1 + 8 * 2 * 32 * 8);
  pack_b_frags<<<16, 256, 0, stream>>>(W1_1, 64,  0,  4, 4, 1.f,   packB1);

  embed_kernel<<<(N_VERT * 64 + 255) / 256, 256, 0, stream>>>(x, embed, F0, Sbf);

  const int ntiles      = N_VERT / 16;            // 3125
  const int expand_grid = (ntiles + 7) / 8;       // 391

  // layer 0
  gemm_expand<<<expand_grid, 256, 0, stream>>>(Sbf, packA0, Gt, Gb, N_VERT);
  ccn_layer_kernel<<<ntiles, 256, 0, stream>>>(rf, Gt, Gb, packB0, s0, Sbf); // Sbf <- bf16(s0)
  // layer 1
  gemm_expand<<<expand_grid, 256, 0, stream>>>(Sbf, packA1, Gt, Gb, N_VERT);
  ccn_layer_kernel<<<ntiles, 256, 0, stream>>>(rf, Gt, Gb, packB1, s1, Sbf);

  mol_kernel<<<N_MOLS, 192, 0, stream>>>(F0, s0, s1, fc_w, fc_b, out);
}
